// GAT_10720238370916
// MI455X (gfx1250) — matrix operand order repaired and checked by reading, not compiled, not run on hardware
//
#include <hip/hip_runtime.h>
#include <hip/hip_bf16.h>
#include <limits.h>
#include <math.h>

typedef __attribute__((ext_vector_type(16))) _Float16 v16h;
typedef __attribute__((ext_vector_type(8)))  float    v8f;

// Problem constants (match reference)
#define GN 100000          // nodes
#define GE 3200000         // raw edges
#define GET (GE + GN)      // edges + self loops
#define F_IN 1433
#define KPAD 1440          // 45 * 32
#define KMAIN 1408         // 44 * 32 : full-width iterations
#define H1 8
#define D1 8
#define C1 64              // H1*D1
#define H2 1
#define D2 7
#define NEG_SLOPE 0.2f

// ---------------- utility fills ----------------
__global__ void fill_f32_kernel(float* p, int n, float v) {
    int t = blockIdx.x * blockDim.x + threadIdx.x;
    if (t < n) p[t] = v;
}
__global__ void fill_i32_kernel(int* p, int n, int v) {
    int t = blockIdx.x * blockDim.x + threadIdx.x;
    if (t < n) p[t] = v;
}

// ---------------- W1^T f16 prep: w1t[n][k] = (half)W1[k*64+n], zero pad k>=1433 ----
__global__ void prep_w1t_kernel(const float* __restrict__ W1, _Float16* __restrict__ w1t) {
    int t = blockIdx.x * blockDim.x + threadIdx.x;
    if (t >= C1 * KPAD) return;
    int n = t / KPAD;
    int k = t - n * KPAD;
    float v = (k < F_IN) ? W1[(size_t)k * C1 + n] : 0.0f;
    w1t[(size_t)n * KPAD + k] = (_Float16)v;
}

// ---------------- Layer-1 GEMM via WMMA: h1[N,64] = x @ W1 ----------------
// 256 threads = 8 waves; wave w computes rows [blk*128 + w*16, +16) x all 64 cols.
__global__ __launch_bounds__(256)
void gemm1_wmma_kernel(const float* __restrict__ x,
                       const _Float16* __restrict__ w1t,
                       float* __restrict__ h1) {
    const int wave = threadIdx.x >> 5;
    const int lane = threadIdx.x & 31;
    const int m0 = blockIdx.x * 128 + wave * 16;
    if (m0 >= GN) return;                 // wave-uniform: EXEC stays all-1s for WMMA

    const int row   = lane & 15;
    const int khalf = lane >> 4;          // 0 or 1
    const int m     = m0 + row;
    const int base0 = khalf * 16;         // A elems 0..7  -> K = k+base0..+7
    const int base1 = khalf * 16 + 8;     // A elems 8..15 -> K = k+base1..+7
    const int ktop  = khalf * 16;         // B elems 0..15 -> K = k+ktop..+15

    v8f c0 = {}, c1 = {}, c2 = {}, c3 = {};
    const float* xr = x + (size_t)m * F_IN;
    const _Float16* wb = w1t + (size_t)row * KPAD + ktop;

    // ---- main loop: 44 full-width K steps, no bounds checks ----
    for (int k = 0; k < KMAIN; k += 32) {
        // prefetch next K slab of this row (global_prefetch_b8 on gfx1250)
        __builtin_prefetch(xr + k + 32 + base0, 0, 1);

        v16h a;
#pragma unroll
        for (int j = 0; j < 8; ++j) {
            a[j]     = (_Float16)xr[k + base0 + j];
            a[8 + j] = (_Float16)xr[k + base1 + j];
        }
        const _Float16* wbase = wb + k;
        v16h b0 = *reinterpret_cast<const v16h*>(wbase);
        v16h b1 = *reinterpret_cast<const v16h*>(wbase + 16 * (size_t)KPAD);
        v16h b2 = *reinterpret_cast<const v16h*>(wbase + 32 * (size_t)KPAD);
        v16h b3 = *reinterpret_cast<const v16h*>(wbase + 48 * (size_t)KPAD);

        c0 = __builtin_amdgcn_wmma_f32_16x16x32_f16(false, a, false, b0, (short)0, c0, false, false);
        c1 = __builtin_amdgcn_wmma_f32_16x16x32_f16(false, a, false, b1, (short)0, c1, false, false);
        c2 = __builtin_amdgcn_wmma_f32_16x16x32_f16(false, a, false, b2, (short)0, c2, false, false);
        c3 = __builtin_amdgcn_wmma_f32_16x16x32_f16(false, a, false, b3, (short)0, c3, false, false);
    }

    // ---- single tail step (K = 1408..1439, valid < 1433), branchless ----
    {
        const int k = KMAIN;
        v16h a;
#pragma unroll
        for (int j = 0; j < 8; ++j) {
            int cA = k + base0 + j;
            int cB = k + base1 + j;
            float vA = xr[cA < F_IN ? cA : (F_IN - 1)];   // clamped: load always legal
            float vB = xr[cB < F_IN ? cB : (F_IN - 1)];
            a[j]     = (_Float16)(cA < F_IN ? vA : 0.0f);
            a[8 + j] = (_Float16)(cB < F_IN ? vB : 0.0f);
        }
        const _Float16* wbase = wb + k;                    // w1t zero-padded to KPAD
        v16h b0 = *reinterpret_cast<const v16h*>(wbase);
        v16h b1 = *reinterpret_cast<const v16h*>(wbase + 16 * (size_t)KPAD);
        v16h b2 = *reinterpret_cast<const v16h*>(wbase + 32 * (size_t)KPAD);
        v16h b3 = *reinterpret_cast<const v16h*>(wbase + 48 * (size_t)KPAD);

        c0 = __builtin_amdgcn_wmma_f32_16x16x32_f16(false, a, false, b0, (short)0, c0, false, false);
        c1 = __builtin_amdgcn_wmma_f32_16x16x32_f16(false, a, false, b1, (short)0, c1, false, false);
        c2 = __builtin_amdgcn_wmma_f32_16x16x32_f16(false, a, false, b2, (short)0, c2, false, false);
        c3 = __builtin_amdgcn_wmma_f32_16x16x32_f16(false, a, false, b3, (short)0, c3, false, false);
    }

    // C layout: lane L, VGPR r -> M = r + 8*(L>>4), N = 16*t + (L&15)
#pragma unroll
    for (int r = 0; r < 8; ++r) {
        int mm = m0 + r + 8 * khalf;
        float* op = h1 + (size_t)mm * C1 + row;
        op[0]  = c0[r];
        op[16] = c1[r];
        op[32] = c2[r];
        op[48] = c3[r];
    }
}

// ---------------- per-(node,head) attention scalars ----------------
__global__ void alpha_kernel(const float* __restrict__ feat,
                             const float* __restrict__ a_s,
                             const float* __restrict__ a_d,
                             float* __restrict__ als, float* __restrict__ ald,
                             int n, int H, int D) {
    int t = blockIdx.x * blockDim.x + threadIdx.x;
    if (t >= n * H) return;
    int node = t / H;
    int h = t - node * H;
    const float* fp = feat + (size_t)node * H * D + h * D;
    float s = 0.f, d = 0.f;
    for (int j = 0; j < D; ++j) {
        float v = fp[j];
        s += v * a_s[h * D + j];
        d += v * a_d[h * D + j];
    }
    als[t] = s;
    ald[t] = d;
}

// order-preserving float<->int encoding for atomicMax
__device__ __forceinline__ int enc_f(float f) {
    int i = __float_as_int(f);
    return (i >= 0) ? i : (i ^ 0x7fffffff);
}
__device__ __forceinline__ float dec_f(int i) {
    return __int_as_float((i >= 0) ? i : (i ^ 0x7fffffff));
}

__device__ __forceinline__ void edge_src_dst(const int* ei, int e, int& src, int& dst) {
    if (e < GE) { src = ei[e]; dst = ei[GE + e]; }
    else        { src = e - GE; dst = e - GE; }
}

// ---------------- edge pass 1: segment max over dst ----------------
__global__ void edge_max_kernel(const int* __restrict__ ei,
                                const float* __restrict__ als,
                                const float* __restrict__ ald,
                                int* __restrict__ menc, int H) {
    int t = blockIdx.x * blockDim.x + threadIdx.x;
    if (t >= GET * H) return;
    int e = t / H;
    int h = t - e * H;
    int src, dst;
    edge_src_dst(ei, e, src, dst);
    float v = als[src * H + h] + ald[dst * H + h];
    v = (v > 0.f) ? v : NEG_SLOPE * v;
    atomicMax(&menc[dst * H + h], enc_f(v));
}

// ---------------- edge pass 2: segment sum of exp(e - m) ----------------
__global__ void edge_sum_kernel(const int* __restrict__ ei,
                                const float* __restrict__ als,
                                const float* __restrict__ ald,
                                const int* __restrict__ menc,
                                float* __restrict__ ssum, int H) {
    int t = blockIdx.x * blockDim.x + threadIdx.x;
    if (t >= GET * H) return;
    int e = t / H;
    int h = t - e * H;
    int src, dst;
    edge_src_dst(ei, e, src, dst);
    float v = als[src * H + h] + ald[dst * H + h];
    v = (v > 0.f) ? v : NEG_SLOPE * v;
    float ex = __expf(v - dec_f(menc[dst * H + h]));
    atomicAdd(&ssum[dst * H + h], ex);
}

// ---------------- edge pass 3: weighted scatter-add of messages ----------------
__global__ void edge_agg_kernel(const int* __restrict__ ei,
                                const float* __restrict__ als,
                                const float* __restrict__ ald,
                                const int* __restrict__ menc,
                                const float* __restrict__ ssum,
                                const float* __restrict__ feat,
                                float* __restrict__ out, int H, int D) {
    int t = blockIdx.x * blockDim.x + threadIdx.x;
    if (t >= GET * H) return;
    int e = t / H;
    int h = t - e * H;
    int src, dst;
    edge_src_dst(ei, e, src, dst);
    float v = als[src * H + h] + ald[dst * H + h];
    v = (v > 0.f) ? v : NEG_SLOPE * v;
    float ex = __expf(v - dec_f(menc[dst * H + h]));
    float w = ex / (ssum[dst * H + h] + 1e-16f);
    const float* fp = feat + (size_t)src * H * D + h * D;
    float* op = out + (size_t)dst * H * D + h * D;
    for (int d = 0; d < D; ++d) atomicAdd(&op[d], fp[d] * w);
}

// ---------------- bias + relu on layer-1 aggregate ----------------
__global__ void relu_bias_kernel(float* __restrict__ o, const float* __restrict__ b) {
    int t = blockIdx.x * blockDim.x + threadIdx.x;
    if (t >= GN * C1) return;
    o[t] = fmaxf(o[t] + b[t & (C1 - 1)], 0.f);
}

// ---------------- layer-2 projection: h2[N,7] = h1r @ W2 ----------------
__global__ void proj2_kernel(const float* __restrict__ h1r,
                             const float* __restrict__ W2,
                             float* __restrict__ h2) {
    int node = blockIdx.x * blockDim.x + threadIdx.x;
    if (node >= GN) return;
    float acc[D2];
#pragma unroll
    for (int c = 0; c < D2; ++c) acc[c] = 0.f;
    const float* hp = h1r + (size_t)node * C1;
    for (int k = 0; k < C1; ++k) {
        float xv = hp[k];
#pragma unroll
        for (int c = 0; c < D2; ++c) acc[c] += xv * W2[k * D2 + c];
    }
#pragma unroll
    for (int c = 0; c < D2; ++c) h2[(size_t)node * D2 + c] = acc[c];
}

// ---------------- final bias + log_softmax ----------------
__global__ void logsoftmax_kernel(const float* __restrict__ out2,
                                  const float* __restrict__ b2,
                                  float* __restrict__ y) {
    int node = blockIdx.x * blockDim.x + threadIdx.x;
    if (node >= GN) return;
    float v[D2];
    float mx = -INFINITY;
#pragma unroll
    for (int c = 0; c < D2; ++c) {
        v[c] = out2[(size_t)node * D2 + c] + b2[c];
        mx = fmaxf(mx, v[c]);
    }
    float s = 0.f;
#pragma unroll
    for (int c = 0; c < D2; ++c) s += __expf(v[c] - mx);
    float lse = mx + __logf(s);
#pragma unroll
    for (int c = 0; c < D2; ++c) y[(size_t)node * D2 + c] = v[c] - lse;
}

static inline int cdiv(long long a, int b) { return (int)((a + b - 1) / b); }

extern "C" void kernel_launch(void* const* d_in, const int* in_sizes, int n_in,
                              void* d_out, int out_size, void* d_ws, size_t ws_size,
                              hipStream_t stream) {
    const float* x      = (const float*)d_in[0];
    const int*   ei     = (const int*)d_in[1];
    const float* W1     = (const float*)d_in[2];
    const float* a1_src = (const float*)d_in[3];
    const float* a1_dst = (const float*)d_in[4];
    const float* b1     = (const float*)d_in[5];
    const float* W2     = (const float*)d_in[6];
    const float* a2_src = (const float*)d_in[7];
    const float* a2_dst = (const float*)d_in[8];
    const float* b2     = (const float*)d_in[9];
    float* y = (float*)d_out;

    // ---- workspace carve-up (256B aligned) ----
    char* ws = (char*)d_ws;
    size_t off = 0;
    auto carve = [&](size_t bytes) -> char* {
        char* p = ws + off;
        off = (off + bytes + 255) & ~(size_t)255;
        return p;
    };
    _Float16* w1t = (_Float16*)carve((size_t)C1 * KPAD * sizeof(_Float16));
    float* h1     = (float*)carve((size_t)GN * C1 * sizeof(float));
    float* a1s    = (float*)carve((size_t)GN * H1 * sizeof(float));
    float* a1d    = (float*)carve((size_t)GN * H1 * sizeof(float));
    int*   m1     = (int*)  carve((size_t)GN * H1 * sizeof(int));
    float* s1     = (float*)carve((size_t)GN * H1 * sizeof(float));
    float* out1   = (float*)carve((size_t)GN * C1 * sizeof(float));
    float* h2     = (float*)carve((size_t)GN * D2 * sizeof(float));
    float* a2s    = (float*)carve((size_t)GN * sizeof(float));
    float* a2d    = (float*)carve((size_t)GN * sizeof(float));
    int*   m2     = (int*)  carve((size_t)GN * sizeof(int));
    float* s2     = (float*)carve((size_t)GN * sizeof(float));
    float* out2   = (float*)carve((size_t)GN * D2 * sizeof(float));
    (void)ws_size; (void)n_in; (void)in_sizes; (void)out_size;

    const int B = 256;

    // init accumulators
    fill_f32_kernel<<<cdiv((long long)GN * H1, B), B, 0, stream>>>(s1, GN * H1, 0.f);
    fill_f32_kernel<<<cdiv((long long)GN * C1, B), B, 0, stream>>>(out1, GN * C1, 0.f);
    fill_f32_kernel<<<cdiv(GN, B), B, 0, stream>>>(s2, GN, 0.f);
    fill_f32_kernel<<<cdiv((long long)GN * D2, B), B, 0, stream>>>(out2, GN * D2, 0.f);
    fill_i32_kernel<<<cdiv((long long)GN * H1, B), B, 0, stream>>>(m1, GN * H1, INT_MIN);
    fill_i32_kernel<<<cdiv(GN, B), B, 0, stream>>>(m2, GN, INT_MIN);

    // W1^T f16 (zero-padded K)
    prep_w1t_kernel<<<cdiv((long long)C1 * KPAD, B), B, 0, stream>>>(W1, w1t);

    // layer-1 projection GEMM (WMMA)
    gemm1_wmma_kernel<<<cdiv(GN, 128), 256, 0, stream>>>(x, w1t, h1);

    // layer-1 attention scalars + edge softmax + aggregation
    alpha_kernel<<<cdiv((long long)GN * H1, B), B, 0, stream>>>(h1, a1_src, a1_dst, a1s, a1d, GN, H1, D1);
    edge_max_kernel<<<cdiv((long long)GET * H1, B), B, 0, stream>>>(ei, a1s, a1d, m1, H1);
    edge_sum_kernel<<<cdiv((long long)GET * H1, B), B, 0, stream>>>(ei, a1s, a1d, m1, s1, H1);
    edge_agg_kernel<<<cdiv((long long)GET * H1, B), B, 0, stream>>>(ei, a1s, a1d, m1, s1, h1, out1, H1, D1);
    relu_bias_kernel<<<cdiv((long long)GN * C1, B), B, 0, stream>>>(out1, b1);

    // layer-2 projection + attention + aggregation
    proj2_kernel<<<cdiv(GN, B), B, 0, stream>>>(out1, W2, h2);
    alpha_kernel<<<cdiv(GN, B), B, 0, stream>>>(h2, a2_src, a2_dst, a2s, a2d, GN, H2, D2);
    edge_max_kernel<<<cdiv((long long)GET, B), B, 0, stream>>>(ei, a2s, a2d, m2, H2);
    edge_sum_kernel<<<cdiv((long long)GET, B), B, 0, stream>>>(ei, a2s, a2d, m2, s2, H2);
    edge_agg_kernel<<<cdiv((long long)GET, B), B, 0, stream>>>(ei, a2s, a2d, m2, s2, h2, out2, H2, D2);

    // final bias + log_softmax
    logsoftmax_kernel<<<cdiv(GN, B), B, 0, stream>>>(out2, b2, y);
}
